// WindowedCrossAttention_4088808865836
// MI455X (gfx1250) — matrix operand order repaired and checked by reading, not compiled, not run on hardware
//
#include <hip/hip_runtime.h>
#include <hip/hip_bf16.h>
#include <cstdint>

typedef __attribute__((ext_vector_type(16))) _Float16 v16h;
typedef __attribute__((ext_vector_type(8)))  _Float16 v8h;
typedef __attribute__((ext_vector_type(8)))  float    v8f;
typedef __attribute__((ext_vector_type(4)))  unsigned int v4u;
typedef __attribute__((ext_vector_type(8)))  int      v8i;
typedef __attribute__((ext_vector_type(4)))  int      v4i;

#define C_DIM  256
#define HEADS  8
#define HD     32
#define S_WIN  256          // 16*16 tokens per window
#define H_IMG  192
#define W_IMG  192
#define HW     (H_IMG * W_IMG)
#define NWH    12
#define NWIN   288          // 2 * 12 * 12
#define WELEM  65536        // C_DIM * S_WIN halves per window

// ---------------------------------------------------------------------------
// CDNA5 async / TDM helpers
// ---------------------------------------------------------------------------
__device__ inline unsigned lds_lo32(const void* p) {
  // flat shared address -> LDS byte offset (aperture truncation, ISA 10.2)
  return (unsigned)(uintptr_t)p;
}

// ASYNCcnt path: global -> LDS, 16 bytes per lane (VDST = LDS address VGPR)
__device__ inline void async_load_b128(unsigned lds_off, const void* gaddr) {
  asm volatile("global_load_async_to_lds_b128 %0, %1, off"
               :: "v"(lds_off), "v"(gaddr)
               : "memory");
}
__device__ inline void async_wait0() {
  asm volatile("s_wait_asynccnt 0x0" ::: "memory");
}

// TENSORcnt path: one TDM descriptor moves a 2-D f16 tile global -> LDS.
// rows x cols elements, row stride = `stride` elements, contiguous in LDS.
__device__ inline void tdm_load_2d_f16(unsigned lds_off, const void* gaddr,
                                       unsigned cols, unsigned rows,
                                       unsigned stride) {
  const unsigned long long ga = (unsigned long long)(uintptr_t)gaddr;
  v4u g0 = { 1u,                                   // count=1, user descriptor
             lds_off,                              // lds_addr  [63:32]
             (unsigned)ga,                         // global_addr[31:0]
             ((unsigned)(ga >> 32) & 0x01FFFFFFu)  // global_addr[56:32]
                 | (2u << 30) };                   // type=2 ("image")
  v8i g1 = { (int)(1u << 16),                      // data_size=1 -> 2 bytes
             (int)(cols << 16),                    // tensor_dim0 [79:48] lo16
             (int)(rows << 16),                    // tensor_dim1 [111:80] lo16
             (int)(cols << 16),                    // tile_dim0   [127:112]
             (int)rows,                            // tile_dim1   [143:128]
             (int)stride,                          // dim0_stride [207:160] lo32
             0, 0 };
  v4i gz4 = { 0, 0, 0, 0 };
  v8i gz8 = { 0, 0, 0, 0, 0, 0, 0, 0 };
  __builtin_amdgcn_tensor_load_to_lds(g0, g1, gz4, gz4, gz8, 0);
}

// ---------------------------------------------------------------------------
// WMMA fragment helpers (CDNA5 ISA 7.12.2 layouts, wave32)
// ---------------------------------------------------------------------------
__device__ inline v16h frag_a(const _Float16* base, int row, int stride, int lane) {
  const int kb = (lane >> 4) << 3;
  const _Float16* p = base + row * stride;
  v8h lo = *(const v8h*)(p + kb);
  v8h hi = *(const v8h*)(p + kb + 16);
  return __builtin_shufflevector(lo, hi, 0,1,2,3,4,5,6,7,8,9,10,11,12,13,14,15);
}

__device__ inline v16h frag_bT(const _Float16* base, int ncol, int lane) {
  const int kb = (lane >> 4) << 3;
  const _Float16* p = base + ncol * 32;
  v8h lo = *(const v8h*)(p + kb);
  v8h hi = *(const v8h*)(p + kb + 16);
  return __builtin_shufflevector(lo, hi, 0,1,2,3,4,5,6,7,8,9,10,11,12,13,14,15);
}

__device__ inline v8f wmma_f16(v16h a, v16h b, v8f c) {
  return __builtin_amdgcn_wmma_f32_16x16x32_f16(false, a, false, b, (short)0, c,
                                                false, false);
}

// ---------------------------------------------------------------------------
// Kernel 1: convert the four 256x256 weight matrices to f16 (row-major [d][c])
// ---------------------------------------------------------------------------
__global__ __launch_bounds__(256) void cvt_w_kernel(const float* __restrict__ Wq,
                                                    const float* __restrict__ Wk,
                                                    const float* __restrict__ Wv,
                                                    const float* __restrict__ Wo,
                                                    _Float16* __restrict__ dst) {
  const int i = blockIdx.x * 256 + threadIdx.x;   // 0 .. 65535
  dst[i]           = (_Float16)Wq[i];
  dst[WELEM + i]   = (_Float16)Wk[i];
  dst[2*WELEM + i] = (_Float16)Wv[i];
  dst[3*WELEM + i] = (_Float16)Wo[i];
}

// ---------------------------------------------------------------------------
// Kernel 2/5: 1x1-conv projection as GEMM  out[d][s] = sum_c W[d][c] * X[c][s]
//  One block = one window x one 128-row d-tile x one 128-col s-tile.
//  8 waves arranged 4(M) x 2(N); each wave -> 32x64 outputs (2x4 WMMA tiles).
//  A tile (f16 weights) staged via global_load_async_to_lds_b128 (ASYNCcnt).
// ---------------------------------------------------------------------------
template <bool IN16, bool OUT16>
__global__ __launch_bounds__(256) void proj_kernel(const float*    __restrict__ x32,
                                                   const _Float16* __restrict__ x16,
                                                   const _Float16* __restrict__ W16,
                                                   const float*    __restrict__ bias,
                                                   _Float16*       __restrict__ out16,
                                                   float*          __restrict__ out32) {
  __shared__ _Float16 As[128 * 32];   // A tile   [M=d 128][K=c 32]
  __shared__ _Float16 Bs[128 * 32];   // B^T tile [N=s 128][K=c 32]

  const int blk  = blockIdx.x;
  const int win  = blk >> 2;
  const int dt   = (blk >> 1) & 1;          // d-tile (0/1)
  const int st   = blk & 1;                 // s-tile (0/1)
  const int b    = win / (NWH * NWH);
  const int wrem = win % (NWH * NWH);
  const int hwin = wrem / NWH;
  const int wwin = wrem % NWH;

  const int tid  = threadIdx.x;
  const int lane = tid & 31;
  const int wid  = tid >> 5;
  const int wm   = wid >> 1;                // 0..3
  const int wn   = wid & 1;                 // 0..1

  v8f acc[2][4];
  const v8f vzero = {0.f,0.f,0.f,0.f,0.f,0.f,0.f,0.f};
#pragma unroll
  for (int mi = 0; mi < 2; ++mi)
#pragma unroll
    for (int ni = 0; ni < 4; ++ni) acc[mi][ni] = vzero;

  // staging assignments
  const int ar   = tid >> 1;                // A row 0..127
  const int ac   = (tid & 1) << 4;          // A col group 0/16
  const int bs_  = tid & 127;               // B s-row 0..127
  const int bc0  = (tid >> 7) << 4;         // B c group 0/16
  const int s_g  = st * 128 + bs_;          // token in window 0..255
  const int srow = s_g >> 4, scol = s_g & 15;
  const int pix  = (hwin * 16 + srow) * W_IMG + wwin * 16 + scol;
  const unsigned as_base = lds_lo32(As) + (unsigned)(ar * 32 + ac) * 2u;

  for (int kk = 0; kk < 8; ++kk) {          // K = 256 in steps of 32
    // ---- stage A: f16 weights, async global->LDS (2 x 16B per thread)
    const _Float16* wsrc = W16 + (size_t)(dt * 128 + ar) * 256 + kk * 32 + ac;
    async_load_b128(as_base,      wsrc);
    async_load_b128(as_base + 16, wsrc + 8);

    // ---- stage B^T: 16 c-strided reads, contiguous 16-half LDS row write
    if (IN16) {
      const _Float16* xp = x16 + (size_t)win * WELEM + (size_t)(kk * 32 + bc0) * S_WIN + s_g;
#pragma unroll
      for (int j = 0; j < 16; ++j) Bs[bs_ * 32 + bc0 + j] = xp[j * S_WIN];
      if (kk < 7) __builtin_prefetch(xp + 32 * S_WIN, 0, 3);
    } else {
      const float* xp = x32 + (size_t)(b * C_DIM + kk * 32 + bc0) * HW + pix;
#pragma unroll
      for (int j = 0; j < 16; ++j) Bs[bs_ * 32 + bc0 + j] = (_Float16)xp[(size_t)j * HW];
      if (kk < 7) __builtin_prefetch(xp + (size_t)32 * HW, 0, 3);
    }
    async_wait0();
    __syncthreads();

    v16h af[2], bf[4];
#pragma unroll
    for (int mi = 0; mi < 2; ++mi)
      af[mi] = frag_a(As, wm * 32 + mi * 16 + (lane & 15), 32, lane);
#pragma unroll
    for (int ni = 0; ni < 4; ++ni)
      bf[ni] = frag_bT(Bs, wn * 64 + ni * 16 + (lane & 15), lane);

#pragma unroll
    for (int mi = 0; mi < 2; ++mi)
#pragma unroll
      for (int ni = 0; ni < 4; ++ni)
        acc[mi][ni] = wmma_f16(af[mi], bf[ni], acc[mi][ni]);
    __syncthreads();
  }

  // ---- epilogue: C layout lane(l), vgpr g -> (M = g + 8*(l/16), N = l%16)
#pragma unroll
  for (int mi = 0; mi < 2; ++mi) {
#pragma unroll
    for (int g = 0; g < 8; ++g) {
      const int d  = dt * 128 + wm * 32 + mi * 16 + g + ((lane >> 4) << 3);
      const float bb = bias[d];
#pragma unroll
      for (int ni = 0; ni < 4; ++ni) {
        const int s = st * 128 + wn * 64 + ni * 16 + (lane & 15);
        const float val = acc[mi][ni][g] + bb;
        if (OUT16) {
          out16[(size_t)win * WELEM + (size_t)d * S_WIN + s] = (_Float16)val;
        } else {
          const int h  = hwin * 16 + (s >> 4);
          const int wc = wwin * 16 + (s & 15);
          out32[((size_t)(b * C_DIM + d) * H_IMG + h) * W_IMG + wc] = val;
        }
      }
    }
  }
}

// ---------------------------------------------------------------------------
// Kernel 3: windowed attention, one block per (window, head), all transposed:
//   S^T[t][s] = K · Q^T   (A = K [t][hd], B^T = Q [s][hd])
//   O^T[hd][s]            (A = V [hd][t], B^T = P^T staged [s][t])
//  V tile staged by the Tensor Data Mover (one descriptor, TENSORcnt).
// ---------------------------------------------------------------------------
__global__ __launch_bounds__(256) void attn_kernel(const _Float16* __restrict__ q16,
                                                   const _Float16* __restrict__ k16,
                                                   const _Float16* __restrict__ v16,
                                                   _Float16*       __restrict__ o16) {
  __shared__ _Float16 Qs[S_WIN * HD];       // [s][hd]
  __shared__ _Float16 Ks[S_WIN * HD];       // [t][hd]
  __shared__ _Float16 Vs[HD * S_WIN];       // [hd][t]
  __shared__ _Float16 Ps[8 * 32 * 32];      // per-wave P^T tiles, [s 32][t 32]

  const int blk  = blockIdx.x;
  const int win  = blk >> 3;
  const int head = blk & 7;
  const int tid  = threadIdx.x;
  const int lane = tid & 31;
  const int wid  = tid >> 5;
  const int s0   = wid * 32;                // this wave's 32 query columns

  const _Float16* qb = q16 + (size_t)win * WELEM + (size_t)head * HD * S_WIN;
  const _Float16* kb = k16 + (size_t)win * WELEM + (size_t)head * HD * S_WIN;
  const _Float16* vb = v16 + (size_t)win * WELEM + (size_t)head * HD * S_WIN;

  // ---- V tile via TDM: 32 rows x 256 cols f16, contiguous -> Vs [hd][t]
  if (wid == 0) {                           // TDM ignores EXEC; issue once
    tdm_load_2d_f16(lds_lo32(Vs), vb, S_WIN, HD, S_WIN);
  }

  // ---- stage Q,K transposed ([hd][s] -> [s][hd]); coalesced per hd plane
#pragma unroll
  for (int hd = 0; hd < HD; ++hd) {
    Qs[tid * HD + hd] = qb[hd * S_WIN + tid];
    Ks[tid * HD + hd] = kb[hd * S_WIN + tid];
  }
  __builtin_amdgcn_s_wait_tensorcnt(0);
  __syncthreads();

  // Q B^T-fragments: reused across all t-tiles (full K = hd = 32)
  v16h bq[2];
#pragma unroll
  for (int sni = 0; sni < 2; ++sni)
    bq[sni] = frag_bT(Qs, s0 + sni * 16 + (lane & 15), lane);

  const v8f vzero = {0.f,0.f,0.f,0.f,0.f,0.f,0.f,0.f};
  v8f ovT[2][2];                            // O^T accum: [hd-tile][s-tile]
#pragma unroll
  for (int h = 0; h < 2; ++h) { ovT[h][0] = vzero; ovT[h][1] = vzero; }
  float m_run[2] = { -1e30f, -1e30f };      // per owned s column (2 per lane)
  float d_run[2] = { 0.f, 0.f };

  const float scale = 0.17677669529663687f; // 1/sqrt(32)
  _Float16* myP = Ps + wid * 1024;

  for (int tt = 0; tt < 8; ++tt) {
    // ---- S^T = K Q^T : tile 32(t) x 32(s)
    v16h ak[2];
#pragma unroll
    for (int tmi = 0; tmi < 2; ++tmi)
      ak[tmi] = frag_a(Ks, tt * 32 + tmi * 16 + (lane & 15), 32, lane);

    v8f sc[2][2];
#pragma unroll
    for (int tmi = 0; tmi < 2; ++tmi)
#pragma unroll
      for (int sni = 0; sni < 2; ++sni)
        sc[tmi][sni] = wmma_f16(ak[tmi], bq[sni], vzero);

    // ---- online softmax per owned column s (lane-local over t)
#pragma unroll
    for (int sni = 0; sni < 2; ++sni) {
#pragma unroll
      for (int tmi = 0; tmi < 2; ++tmi)
#pragma unroll
        for (int g = 0; g < 8; ++g) sc[tmi][sni][g] *= scale;

      float tm = sc[0][sni][0];
#pragma unroll
      for (int g = 1; g < 8; ++g) tm = fmaxf(tm, sc[0][sni][g]);
#pragma unroll
      for (int g = 0; g < 8; ++g) tm = fmaxf(tm, sc[1][sni][g]);
      tm = fmaxf(tm, __shfl_xor(tm, 16, 32));          // join the two halves

      const float mn   = fmaxf(m_run[sni], tm);
      const float corr = __expf(m_run[sni] - mn);
      m_run[sni] = mn;

      float rs = 0.f;
#pragma unroll
      for (int tmi = 0; tmi < 2; ++tmi) {
        v8h pk;
#pragma unroll
        for (int g = 0; g < 8; ++g) {
          const float pv = __expf(sc[tmi][sni][g] - mn);
          rs += pv;
          pk[g] = (_Float16)pv;
        }
        // P^T C-layout: t = tmi*16 + 8*(lane/16) + g  -> 8 contiguous t
        *(v8h*)(&myP[(sni * 16 + (lane & 15)) * 32 + tmi * 16 +
                     ((lane >> 4) << 3)]) = pk;
      }
      rs += __shfl_xor(rs, 16, 32);
      d_run[sni] = d_run[sni] * corr + rs;
#pragma unroll
      for (int h = 0; h < 2; ++h)
#pragma unroll
        for (int g = 0; g < 8; ++g) ovT[h][sni][g] *= corr;
    }

    // ---- O^T += (V tile) x (P^T tile)   (same-wave DS store->load in order)
    v16h av[2], bp[2];
#pragma unroll
    for (int hmi = 0; hmi < 2; ++hmi)
      av[hmi] = frag_a(Vs + tt * 32, hmi * 16 + (lane & 15), S_WIN, lane);
#pragma unroll
    for (int sni = 0; sni < 2; ++sni)
      bp[sni] = frag_bT(myP, sni * 16 + (lane & 15), lane);
#pragma unroll
    for (int hmi = 0; hmi < 2; ++hmi)
#pragma unroll
      for (int sni = 0; sni < 2; ++sni)
        ovT[hmi][sni] = wmma_f16(av[hmi], bp[sni], ovT[hmi][sni]);
  }

  // ---- normalize (per-column scalar) and write o16[win][head*32+hd][s]
#pragma unroll
  for (int sni = 0; sni < 2; ++sni) {
    const float rcp = 1.0f / d_run[sni];
    const int s = s0 + sni * 16 + (lane & 15);
#pragma unroll
    for (int hmi = 0; hmi < 2; ++hmi) {
#pragma unroll
      for (int g = 0; g < 8; ++g) {
        const int hd = hmi * 16 + g + ((lane >> 4) << 3);
        o16[(size_t)win * WELEM + (size_t)(head * HD + hd) * S_WIN + s] =
            (_Float16)(ovT[hmi][sni][g] * rcp);
      }
    }
  }
}

// ---------------------------------------------------------------------------
// Launcher
// ---------------------------------------------------------------------------
extern "C" void kernel_launch(void* const* d_in, const int* in_sizes, int n_in,
                              void* d_out, int out_size, void* d_ws, size_t ws_size,
                              hipStream_t stream) {
  (void)in_sizes; (void)n_in; (void)out_size; (void)ws_size;

  const float* x_q  = (const float*)d_in[0];
  const float* x_kv = (const float*)d_in[1];
  const float* Wq   = (const float*)d_in[2];
  const float* bq   = (const float*)d_in[3];
  const float* Wk   = (const float*)d_in[4];
  const float* bk   = (const float*)d_in[5];
  const float* Wv   = (const float*)d_in[6];
  const float* bv   = (const float*)d_in[7];
  const float* Wo   = (const float*)d_in[8];
  const float* bo   = (const float*)d_in[9];
  float* out = (float*)d_out;

  // workspace layout (f16): 4 weights + Q + K + V + attnO, ~152 MB
  _Float16* wsh = (_Float16*)d_ws;
  _Float16* W16 = wsh;                                   // 4 * 65536
  _Float16* q16 = wsh + 4 * (size_t)WELEM;
  _Float16* k16 = q16 + (size_t)NWIN * WELEM;
  _Float16* v16 = k16 + (size_t)NWIN * WELEM;
  _Float16* o16 = v16 + (size_t)NWIN * WELEM;

  cvt_w_kernel<<<256, 256, 0, stream>>>(Wq, Wk, Wv, Wo, W16);

  const dim3 pg(NWIN * 4);   // win x 2 d-tiles x 2 s-tiles
  proj_kernel<false, true><<<pg, 256, 0, stream>>>(x_q,  nullptr, W16,             bq, q16, nullptr);
  proj_kernel<false, true><<<pg, 256, 0, stream>>>(x_kv, nullptr, W16 + WELEM,     bk, k16, nullptr);
  proj_kernel<false, true><<<pg, 256, 0, stream>>>(x_kv, nullptr, W16 + 2 * WELEM, bv, v16, nullptr);

  attn_kernel<<<NWIN * HEADS, 256, 0, stream>>>(q16, k16, v16, o16);

  proj_kernel<true, false><<<pg, 256, 0, stream>>>(nullptr, o16, W16 + 3 * WELEM, bo, nullptr, out);
}
